// CMPNN_80685255622667
// MI455X (gfx1250) — compile-verified, matrix operand order checked by reading, not run on hardware
//
#include <hip/hip_runtime.h>

// ---------------------------------------------------------------------------
// CMPNN on gfx1250: bf16 WMMA (v_wmma_f32_16x16x32_bf16) for all dense GEMMs.
// h lives at d_out[0 : N*128), e at d_out[N*128 : N*128 + E*128).
// ---------------------------------------------------------------------------

typedef __attribute__((ext_vector_type(16))) __bf16 v16bf;
typedef __attribute__((ext_vector_type(8)))  float  v8f;
typedef unsigned short bf16u;   // raw bf16 storage for weight buffers

static constexpr int DIM = 128;   // hidden dim
static constexpr int WPB = 2;     // waves per block (64 threads)

// ---------------- small device helpers ----------------

__device__ inline v8f wmma_bf16(v16bf a, v16bf b, v8f c) {
  return __builtin_amdgcn_wmma_f32_16x16x32_bf16(
      /*neg_a=*/false, a, /*neg_b=*/false, b,
      /*c_mod=*/(short)0, c, /*reuse_a=*/false, /*reuse_b=*/false);
}

__device__ inline v8f vzero8() {
  v8f z = {0.f, 0.f, 0.f, 0.f, 0.f, 0.f, 0.f, 0.f};
  return z;
}

__device__ inline unsigned f2bf(float f) {  // RNE f32->bf16 (weight prep)
  unsigned u = __float_as_uint(f);
  return (u + 0x7FFFu + ((u >> 16) & 1u)) >> 16;
}

__device__ inline unsigned fkey(float f) {  // monotonic float->uint key
  unsigned u = __float_as_uint(f);
  return (u & 0x80000000u) ? ~u : (u | 0x80000000u);
}
__device__ inline float kinv(unsigned k) {
  unsigned u = (k & 0x80000000u) ? (k & 0x7fffffffu) : ~k;
  return __uint_as_float(u);
}

__device__ inline float lrelu(float x, float s) { return x >= 0.f ? x : s * x; }
__device__ inline float sigmoidf_(float x) { return 1.f / (1.f + __expf(-x)); }

union FragU {
  v16bf v;
  unsigned short s[16];
  uint4 q[2];
};

// B fragment from bf16 [N][K] weight; p points at (n_row, k0 + hi*8)
__device__ inline v16bf frag_w(const bf16u* p) {
  FragU u;
  u.q[0] = *(const uint4*)(p);
  u.q[1] = *(const uint4*)(p + 16);
  return u.v;
}

// A fragment from fp32 row; p points at (row, k0 + hi*8). Native packed cvt.
__device__ inline v16bf frag_x(const float* p) {
  v16bf a;
#pragma unroll
  for (int j = 0; j < 8; ++j) {
    a[j]     = (__bf16)p[j];
    a[8 + j] = (__bf16)p[16 + j];
  }
  return a;
}

// store a 16x16 f32 accumulator tile (+column bias) into an LDS [16][128] tile
__device__ inline void store_tile_lds(float* t, v8f acc, const float* bias,
                                      int n, int mr, int hi) {
  int col = n * 16 + mr;
  float b = bias[col];
#pragma unroll
  for (int i = 0; i < 8; ++i) t[(i + 8 * hi) * DIM + col] = acc[i] + b;
}

// dual-tile GEMM pass: A fragments resident, stream B for tiles n and n+1.
// KF = number of 32-wide K fragments. ldk = weight row length (K).
template <int KF>
__device__ inline void gemm_store2(const v16bf* aF, const bf16u* Wt, size_t ldk,
                                   int n, int mr, int hi, float* tile,
                                   const float* bias) {
  v8f acc0 = vzero8(), acc1 = vzero8();
  const bf16u* w0 = Wt + (size_t)(n * 16 + mr) * ldk + hi * 8;
  const bf16u* w1 = Wt + (size_t)((n + 1) * 16 + mr) * ldk + hi * 8;
#pragma unroll
  for (int kk = 0; kk < KF; ++kk) {
    v16bf b0 = frag_w(w0 + kk * 32);
    v16bf b1 = frag_w(w1 + kk * 32);
    acc0 = wmma_bf16(aF[kk], b0, acc0);
    acc1 = wmma_bf16(aF[kk], b1, acc1);
  }
  store_tile_lds(tile, acc0, bias, n, mr, hi);
  store_tile_lds(tile, acc1, bias, n + 1, mr, hi);
}

// row-wise LayerNorm + LeakyReLU over an LDS [16][128] tile (in place).
// lane (mr,hi) handles row mr, cols [hi*64, hi*64+64).
__device__ inline void ln_lrelu_tile(float* t, const float* g, const float* be,
                                     float slope, int mr, int hi) {
  float s1 = 0.f, s2 = 0.f;
  float* row = t + mr * DIM + hi * 64;
#pragma unroll 8
  for (int c = 0; c < 64; ++c) { float v = row[c]; s1 += v; s2 += v * v; }
  s1 += __shfl_xor(s1, 16);
  s2 += __shfl_xor(s2, 16);
  float m   = s1 * (1.f / 128.f);
  float var = s2 * (1.f / 128.f) - m * m;
  float inv = rsqrtf(var + 1e-5f);
  const float* gg = g + hi * 64;
  const float* bb = be + hi * 64;
#pragma unroll 8
  for (int c = 0; c < 64; ++c) {
    float v = (row[c] - m) * inv * gg[c] + bb[c];
    row[c] = lrelu(v, slope);
  }
}

// GRU + residual + LeakyReLU(0.01), state updated in place.
// tni: LDS [16][128] fp32 input 'i' of the GRU (booster output).
__device__ inline void gru_phase(const float* tni, float* state, int rows,
                                 const bf16u* Wiht, const bf16u* Whht,
                                 const float* bih, const float* bhh,
                                 const bf16u* Wrest, const float* resb,
                                 int base, int mr, int hi) {
  v16bf iA[4], sA[4];
  int rowc = min(base + mr, rows - 1);
#pragma unroll
  for (int kk = 0; kk < 4; ++kk) {
    iA[kk] = frag_x(tni + mr * DIM + kk * 32 + hi * 8);
    sA[kk] = frag_x(state + (size_t)rowc * DIM + kk * 32 + hi * 8);
  }
  for (int j = 0; j < 8; ++j) {
    v8f ir = vzero8(), iz = vzero8(), in_ = vzero8();
    v8f hr = vzero8(), hz = vzero8(), hn = vzero8(), rs = vzero8();
#pragma unroll
    for (int kk = 0; kk < 4; ++kk) {
      int ko = kk * 32 + hi * 8;
      ir = wmma_bf16(iA[kk], frag_w(Wiht + (size_t)(      j * 16 + mr) * DIM + ko), ir);
      iz = wmma_bf16(iA[kk], frag_w(Wiht + (size_t)(128 + j * 16 + mr) * DIM + ko), iz);
      in_= wmma_bf16(iA[kk], frag_w(Wiht + (size_t)(256 + j * 16 + mr) * DIM + ko), in_);
      hr = wmma_bf16(sA[kk], frag_w(Whht + (size_t)(      j * 16 + mr) * DIM + ko), hr);
      hz = wmma_bf16(sA[kk], frag_w(Whht + (size_t)(128 + j * 16 + mr) * DIM + ko), hz);
      hn = wmma_bf16(sA[kk], frag_w(Whht + (size_t)(256 + j * 16 + mr) * DIM + ko), hn);
      rs = wmma_bf16(sA[kk], frag_w(Wrest + (size_t)(     j * 16 + mr) * DIM + ko), rs);
    }
    int col = j * 16 + mr;
    float bir = bih[col], biz = bih[128 + col], bin = bih[256 + col];
    float bhr = bhh[col], bhz = bhh[128 + col], bhn = bhh[256 + col];
    float rb = resb[col];
#pragma unroll
    for (int i = 0; i < 8; ++i) {
      int grow = base + i + 8 * hi;
      int gr = min(grow, rows - 1);
      float old = state[(size_t)gr * DIM + col];
      float r  = sigmoidf_(ir[i] + bir + hr[i] + bhr);
      float z  = sigmoidf_(iz[i] + biz + hz[i] + bhz);
      float nn = tanhf(in_[i] + bin + r * (hn[i] + bhn));
      float v  = (1.f - z) * nn + z * old + rs[i] + rb;
      if (grow < rows) state[(size_t)grow * DIM + col] = lrelu(v, 0.01f);
    }
  }
}

// ---------------- kernels ----------------

// transpose+convert weight: in [K][N] f32 -> out [N][K] bf16
__global__ void tr_kernel(const float* __restrict__ in, bf16u* __restrict__ out,
                          int K, int N, int total) {
  int t = blockIdx.x * blockDim.x + threadIdx.x;
  if (t >= total) return;
  int n = t / K, k = t % K;
  out[t] = (bf16u)f2bf(in[(size_t)k * N + n]);
}

// out = lrelu(LN(X @ W + b), slope)
__global__ void __launch_bounds__(WPB * 32)
encode_kernel(const float* __restrict__ X, const bf16u* __restrict__ Wt,
              const float* __restrict__ b, const float* __restrict__ g,
              const float* __restrict__ be, float* __restrict__ out,
              int rows, float slope) {
  __shared__ float t0[WPB][16 * DIM];
  int wave = threadIdx.x >> 5, lane = threadIdx.x & 31;
  int mr = lane & 15, hi = lane >> 4;
  int base = (blockIdx.x * WPB + wave) * 16;
  int row = min(base + mr, rows - 1);
  v16bf aF[4];
#pragma unroll
  for (int kk = 0; kk < 4; ++kk)
    aF[kk] = frag_x(X + (size_t)row * DIM + kk * 32 + hi * 8);
  for (int n = 0; n < 8; n += 2)
    gemm_store2<4>(aF, Wt, DIM, n, mr, hi, t0[wave], b);
  __syncthreads();
  ln_lrelu_tile(t0[wave], g, be, slope, mr, hi);
  __syncthreads();
  if (base + mr < rows) {
    float* dst = out + (size_t)(base + mr) * DIM + hi * 64;
    const float* sp = t0[wave] + mr * DIM + hi * 64;
#pragma unroll 8
    for (int c = 0; c < 64; ++c) dst[c] = sp[c];
  }
}

__global__ void deg_kernel(const int* __restrict__ dst, int* __restrict__ deg, int E) {
  int i = blockIdx.x * blockDim.x + threadIdx.x;
  if (i < E) atomicAdd(&deg[dst[i]], 1);
}

// edge update: booster(msg) -> GRU(e) + e@eres, in-place on e
__global__ void __launch_bounds__(WPB * 32)
edge_update_kernel(const float* __restrict__ h, float* __restrict__ e,
                   const int* __restrict__ src, const int* __restrict__ dst,
                   const bf16u* __restrict__ W1t, const float* b1,
                   const float* g1, const float* be1,
                   const bf16u* __restrict__ W2t, const float* b2,
                   const bf16u* __restrict__ Wiht, const bf16u* __restrict__ Whht,
                   const float* bih, const float* bhh,
                   const bf16u* __restrict__ Wrest, const float* resb, int E) {
  __shared__ float t0[WPB][16 * DIM];
  __shared__ float t1[WPB][16 * DIM];
  int wave = threadIdx.x >> 5, lane = threadIdx.x & 31;
  int mr = lane & 15, hi = lane >> 4;
  int base = (blockIdx.x * WPB + wave) * 16;
  int row = min(base + mr, E - 1);
  int s = src[row], d2 = dst[row];
  // phase 1: t0 = concat(h[src],h[dst],e) @ W1 + b1   (K = 384)
  v16bf mF[12];
#pragma unroll
  for (int kk = 0; kk < 12; ++kk) {
    int kc = kk * 32 + hi * 8;
    const float* p;
    if (kc < 128)      p = h + (size_t)s * DIM + kc;
    else if (kc < 256) p = h + (size_t)d2 * DIM + (kc - 128);
    else               p = e + (size_t)row * DIM + (kc - 256);
    mF[kk] = frag_x(p);
  }
  for (int n = 0; n < 8; n += 2)
    gemm_store2<12>(mF, W1t, 384, n, mr, hi, t0[wave], b1);
  __syncthreads();
  ln_lrelu_tile(t0[wave], g1, be1, 0.2f, mr, hi);
  __syncthreads();
  // phase 2: t1 = t0 @ W2 + b2
  v16bf tF[4];
#pragma unroll
  for (int kk = 0; kk < 4; ++kk)
    tF[kk] = frag_x(t0[wave] + mr * DIM + kk * 32 + hi * 8);
  for (int n = 0; n < 8; n += 2)
    gemm_store2<4>(tF, W2t, DIM, n, mr, hi, t1[wave], b2);
  __syncthreads();
  gru_phase(t1[wave], e, E, Wiht, Whht, bih, bhh, Wrest, resb, base, mr, hi);
}

// mailbox: ssum += m, smax = max(smax, key(m)) ; m = [h[src], e]
__global__ void aggregate_kernel(const float* __restrict__ h, const float* __restrict__ e,
                                 const int* __restrict__ src, const int* __restrict__ dst,
                                 float* __restrict__ ssum, unsigned* __restrict__ smax,
                                 int E) {
  int t = threadIdx.x;
  int edge = blockIdx.x * 2 + (t >> 7);
  int c = t & 127;
  if (edge >= E) return;
  int s = src[edge], d = dst[edge];
  float v1 = h[(size_t)s * DIM + c];
  float v2 = e[(size_t)edge * DIM + c];
  atomicAdd(&ssum[(size_t)d * 256 + c], v1);
  atomicAdd(&ssum[(size_t)d * 256 + 128 + c], v2);
  atomicMax(&smax[(size_t)d * 256 + c], fkey(v1));
  atomicMax(&smax[(size_t)d * 256 + 128 + c], fkey(v2));
}

// node update: booster([ssum,smax]) -> GRU(h) + h@ares, in-place on h
__global__ void __launch_bounds__(WPB * 32)
node_update_kernel(float* __restrict__ hst, const float* __restrict__ ssum,
                   const unsigned* __restrict__ smax, const int* __restrict__ deg,
                   const bf16u* __restrict__ W1t, const float* b1,
                   const float* g1, const float* be1,
                   const bf16u* __restrict__ W2t, const float* b2,
                   const bf16u* __restrict__ Wiht, const bf16u* __restrict__ Whht,
                   const float* bih, const float* bhh,
                   const bf16u* __restrict__ Wrest, const float* resb, int N) {
  __shared__ float t0[WPB][16 * DIM];
  __shared__ float t1[WPB][16 * DIM];
  int wave = threadIdx.x >> 5, lane = threadIdx.x & 31;
  int mr = lane & 15, hi = lane >> 4;
  int base = (blockIdx.x * WPB + wave) * 16;
  int row = min(base + mr, N - 1);
  int dgr = deg[row];
  // phase 1: t0 = [ssum(256) | smax(256)] @ W1 + b1   (K = 512)
  v16bf aF[16];
#pragma unroll
  for (int kk = 0; kk < 16; ++kk) {
    int kc = kk * 32 + hi * 8;
    if (kc < 256) {
      aF[kk] = frag_x(ssum + (size_t)row * 256 + kc);
    } else {
      const unsigned* q = smax + (size_t)row * 256 + (kc - 256);
      v16bf a;
#pragma unroll
      for (int j = 0; j < 8; ++j) {
        a[j]     = (__bf16)(dgr > 0 ? kinv(q[j]) : 0.f);
        a[8 + j] = (__bf16)(dgr > 0 ? kinv(q[16 + j]) : 0.f);
      }
      aF[kk] = a;
    }
  }
  for (int n = 0; n < 8; n += 2)
    gemm_store2<16>(aF, W1t, 512, n, mr, hi, t0[wave], b1);
  __syncthreads();
  ln_lrelu_tile(t0[wave], g1, be1, 0.2f, mr, hi);
  __syncthreads();
  v16bf tF[4];
#pragma unroll
  for (int kk = 0; kk < 4; ++kk)
    tF[kk] = frag_x(t0[wave] + mr * DIM + kk * 32 + hi * 8);
  for (int n = 0; n < 8; n += 2)
    gemm_store2<4>(tF, W2t, DIM, n, mr, hi, t1[wave], b2);
  __syncthreads();
  gru_phase(t1[wave], hst, N, Wiht, Whht, bih, bhh, Wrest, resb, base, mr, hi);
}

// ---------------- host ----------------

extern "C" void kernel_launch(void* const* d_in, const int* in_sizes, int n_in,
                              void* d_out, int out_size, void* d_ws, size_t ws_size,
                              hipStream_t stream) {
  const int L = 3;
  const float* node_feats = (const float*)d_in[0];
  const float* edge_feats = (const float*)d_in[1];
  const int*   src        = (const int*)d_in[2];
  const int*   dst        = (const int*)d_in[3];
  const float* enc_atom_W = (const float*)d_in[4];
  const float* enc_atom_b = (const float*)d_in[5];
  const float* enc_atom_g = (const float*)d_in[6];
  const float* enc_atom_be= (const float*)d_in[7];
  const float* enc_bond_W = (const float*)d_in[8];
  const float* enc_bond_b = (const float*)d_in[9];
  const float* enc_bond_g = (const float*)d_in[10];
  const float* enc_bond_be= (const float*)d_in[11];
  const float* mb_W1 = (const float*)d_in[12];
  const float* mb_b1 = (const float*)d_in[13];
  const float* mb_g  = (const float*)d_in[14];
  const float* mb_be = (const float*)d_in[15];
  const float* mb_W2 = (const float*)d_in[16];
  const float* mb_b2 = (const float*)d_in[17];
  const float* nb_W1 = (const float*)d_in[18];
  const float* nb_b1 = (const float*)d_in[19];
  const float* nb_g  = (const float*)d_in[20];
  const float* nb_be = (const float*)d_in[21];
  const float* nb_W2 = (const float*)d_in[22];
  const float* nb_b2 = (const float*)d_in[23];
  const float* egru_Wih = (const float*)d_in[24];
  const float* egru_Whh = (const float*)d_in[25];
  const float* egru_bih = (const float*)d_in[26];
  const float* egru_bhh = (const float*)d_in[27];
  const float* agru_Wih = (const float*)d_in[28];
  const float* agru_Whh = (const float*)d_in[29];
  const float* agru_bih = (const float*)d_in[30];
  const float* agru_bhh = (const float*)d_in[31];
  const float* ares_W = (const float*)d_in[32];
  const float* ares_b = (const float*)d_in[33];
  const float* eres_W = (const float*)d_in[34];
  const float* eres_b = (const float*)d_in[35];

  const int N = in_sizes[0] / DIM;
  const int E = in_sizes[2];

  // workspace carve
  char* wp = (char*)d_ws;
  auto carve = [&](size_t bytes) -> char* {
    char* p = wp;
    wp += (bytes + 255) & ~(size_t)255;
    return p;
  };
  float*    ssum = (float*)carve((size_t)N * 256 * 4);
  unsigned* smax = (unsigned*)carve((size_t)N * 256 * 4);
  int*      deg  = (int*)carve((size_t)N * 4);
  bf16u* encAt = (bf16u*)carve((size_t)128 * 128 * 2);
  bf16u* encBt = (bf16u*)carve((size_t)128 * 128 * 2);
  bf16u* mbW1t = (bf16u*)carve((size_t)3 * 128 * 384 * 2);
  bf16u* mbW2t = (bf16u*)carve((size_t)3 * 128 * 128 * 2);
  bf16u* nbW1t = (bf16u*)carve((size_t)3 * 128 * 512 * 2);
  bf16u* nbW2t = (bf16u*)carve((size_t)3 * 128 * 128 * 2);
  bf16u* eWiht = (bf16u*)carve((size_t)3 * 384 * 128 * 2);
  bf16u* eWhht = (bf16u*)carve((size_t)3 * 384 * 128 * 2);
  bf16u* aWiht = (bf16u*)carve((size_t)3 * 384 * 128 * 2);
  bf16u* aWhht = (bf16u*)carve((size_t)3 * 384 * 128 * 2);
  bf16u* arest = (bf16u*)carve((size_t)3 * 128 * 128 * 2);
  bf16u* erest = (bf16u*)carve((size_t)3 * 128 * 128 * 2);

  auto tr = [&](const float* in, bf16u* out, int K, int Nn) {
    int tot = K * Nn;
    tr_kernel<<<(tot + 255) / 256, 256, 0, stream>>>(in, out, K, Nn, tot);
  };
  tr(enc_atom_W, encAt, 128, 128);
  tr(enc_bond_W, encBt, 128, 128);
  for (int l = 0; l < L; ++l) {
    tr(mb_W1 + (size_t)l * 384 * 128, mbW1t + (size_t)l * 128 * 384, 384, 128);
    tr(mb_W2 + (size_t)l * 128 * 128, mbW2t + (size_t)l * 128 * 128, 128, 128);
    tr(nb_W1 + (size_t)l * 512 * 128, nbW1t + (size_t)l * 128 * 512, 512, 128);
    tr(nb_W2 + (size_t)l * 128 * 128, nbW2t + (size_t)l * 128 * 128, 128, 128);
    tr(egru_Wih + (size_t)l * 128 * 384, eWiht + (size_t)l * 384 * 128, 128, 384);
    tr(egru_Whh + (size_t)l * 128 * 384, eWhht + (size_t)l * 384 * 128, 128, 384);
    tr(agru_Wih + (size_t)l * 128 * 384, aWiht + (size_t)l * 384 * 128, 128, 384);
    tr(agru_Whh + (size_t)l * 128 * 384, aWhht + (size_t)l * 384 * 128, 128, 384);
    tr(ares_W + (size_t)l * 128 * 128, arest + (size_t)l * 128 * 128, 128, 128);
    tr(eres_W + (size_t)l * 128 * 128, erest + (size_t)l * 128 * 128, 128, 128);
  }

  float* h = (float*)d_out;            // [N,128]
  float* e = h + (size_t)N * DIM;      // [E,128]

  int nodeBlocks = (N + WPB * 16 - 1) / (WPB * 16);
  int edgeBlocks = (E + WPB * 16 - 1) / (WPB * 16);

  encode_kernel<<<nodeBlocks, WPB * 32, 0, stream>>>(
      node_feats, encAt, enc_atom_b, enc_atom_g, enc_atom_be, h, N, 0.01f);
  encode_kernel<<<edgeBlocks, WPB * 32, 0, stream>>>(
      edge_feats, encBt, enc_bond_b, enc_bond_g, enc_bond_be, e, E, 0.01f);

  hipMemsetAsync(deg, 0, (size_t)N * 4, stream);
  deg_kernel<<<(E + 255) / 256, 256, 0, stream>>>(dst, deg, E);

  for (int l = 0; l < L; ++l) {
    edge_update_kernel<<<edgeBlocks, WPB * 32, 0, stream>>>(
        h, e, src, dst,
        mbW1t + (size_t)l * 128 * 384, mb_b1 + l * 128, mb_g + l * 128, mb_be + l * 128,
        mbW2t + (size_t)l * 128 * 128, mb_b2 + l * 128,
        eWiht + (size_t)l * 384 * 128, eWhht + (size_t)l * 384 * 128,
        egru_bih + l * 384, egru_bhh + l * 384,
        erest + (size_t)l * 128 * 128, eres_b + l * 128, E);

    hipMemsetAsync(ssum, 0, (size_t)N * 256 * 4, stream);
    hipMemsetAsync(smax, 0, (size_t)N * 256 * 4, stream);
    aggregate_kernel<<<(E + 1) / 2, 256, 0, stream>>>(h, e, src, dst, ssum, smax, E);

    node_update_kernel<<<nodeBlocks, WPB * 32, 0, stream>>>(
        h, ssum, smax, deg,
        nbW1t + (size_t)l * 128 * 512, nb_b1 + l * 128, nb_g + l * 128, nb_be + l * 128,
        nbW2t + (size_t)l * 128 * 128, nb_b2 + l * 128,
        aWiht + (size_t)l * 384 * 128, aWhht + (size_t)l * 384 * 128,
        agru_bih + l * 384, agru_bhh + l * 384,
        arest + (size_t)l * 128 * 128, ares_b + l * 128, N);
  }
}
